// JumpAttention_79267916415445
// MI455X (gfx1250) — matrix operand order repaired and checked by reading, not compiled, not run on hardware
//
#include <hip/hip_runtime.h>

// ---------------------------------------------------------------------------
// Problem dims (fixed by reference)
// ---------------------------------------------------------------------------
#define BB 16
#define SS 1024
#define HH 2048
#define DD 256

typedef __attribute__((ext_vector_type(16))) __bf16 v16bf;
typedef __attribute__((ext_vector_type(8)))  float  v8f;

#ifndef __has_builtin
#define __has_builtin(x) 0
#endif
#if __has_builtin(__builtin_amdgcn_global_load_async_to_lds_b128)
#define HAVE_ASYNC 1
#else
#define HAVE_ASYNC 0
#endif

// ---------------------------------------------------------------------------
// Helpers
// ---------------------------------------------------------------------------
static __device__ inline __bf16 f2bf(float f) {
    union { float f; unsigned u; } v; v.f = f;
    unsigned r = v.u + 0x7fffu + ((v.u >> 16) & 1u);   // round-to-nearest-even
    unsigned short h = (unsigned short)(r >> 16);
    return __builtin_bit_cast(__bf16, h);
}

static __device__ inline v8f v8f_zero() {
    v8f z;
    #pragma unroll
    for (int i = 0; i < 8; ++i) z[i] = 0.0f;
    return z;
}

union FragU { v16bf v; uint4 q[2]; };

// A-matrix fragment (16x32 bf16), per ISA table:
//   lane t: row = t&15 ; VGPR0-3 hold k0..k0+7 (k0 = t<16 ? 0 : 8),
//   VGPR4-7 hold 16+k0 .. 16+k0+7
static __device__ inline v16bf load_a_frag(const __bf16* row, int kb, int lane) {
    const int k0 = (lane < 16) ? 0 : 8;
    FragU u;
    u.q[0] = *(const uint4*)(row + kb + k0);
    u.q[1] = *(const uint4*)(row + kb + 16 + k0);
    return u.v;
}

// B-matrix fragment (32x16 bf16): lane t: col n = t&15 ;
//   VGPR0-3 hold k0..k0+7 (k0 = t<16 ? 0 : 8), VGPR4-7 hold 16+k0 .. 16+k0+7.
// `row` points at row n of B^T (i.e. column n of B), contiguous over K.
static __device__ inline v16bf load_b_frag(const __bf16* row, int kb, int lane) {
    const int k0 = (lane < 16) ? 0 : 8;
    const uint4* p = (const uint4*)(row + kb + k0);
    FragU u;
    u.q[0] = p[0];
    u.q[1] = *(const uint4*)(row + kb + 16 + k0);
    return u.v;
}

static __device__ inline v8f wmma_bf16(v16bf a, v16bf b, v8f c) {
    // (neg_a, A, neg_b, B, c_mod, C, reuse_a, reuse_b)
    return __builtin_amdgcn_wmma_f32_16x16x32_bf16(false, a, false, b,
                                                   (short)0, c, false, false);
}

#if HAVE_ASYNC
// Builtin prototype (from compiler diagnostic): pointers to int vector_size(16),
// global side in __device__ (AS1), LDS side in AS3.
typedef int v4i_vs __attribute__((vector_size(16)));
// Wave-wide async DMA of 16B/lane from global into LDS (tracked by ASYNCcnt).
static __device__ inline void async_g2l_b128(const void* g, void* l) {
    __builtin_amdgcn_global_load_async_to_lds_b128(
        (__attribute__((address_space(1))) v4i_vs*)(g),
        (__attribute__((address_space(3))) v4i_vs*)(l),
        0, 0);
}
#endif

// ---------------------------------------------------------------------------
// Elementwise cast / transpose kernels
// ---------------------------------------------------------------------------
__global__ void cast_f32_bf16(const float* __restrict__ src,
                              __bf16* __restrict__ dst, int n) {
    int i = blockIdx.x * blockDim.x + threadIdx.x;
    if (i < n) dst[i] = f2bf(src[i]);
}

// src [R, C] f32  ->  dst [C, R] bf16
__global__ void transpose_cast(const float* __restrict__ src,
                               __bf16* __restrict__ dst, int R, int C) {
    int i = blockIdx.x * blockDim.x + threadIdx.x;
    if (i >= R * C) return;
    int r = i / C, c = i - r * C;
    dst[(size_t)c * R + r] = f2bf(src[i]);
}

// ---------------------------------------------------------------------------
// Projection GEMM:  C[M=B*S, N=D] = A[M, H] * W   (W given as W^T [D, H])
// bf16 in, f32 acc, bf16 out (+ optional per-batch transposed bf16 copy [B,D,S])
// One wave -> 16(M) x 64(N) tile, software-pipelined fragment loads.
// ---------------------------------------------------------------------------
__global__ __launch_bounds__(256) void proj_gemm(const __bf16* __restrict__ A,
                                                 const __bf16* __restrict__ Bt,
                                                 __bf16* __restrict__ Cm,
                                                 __bf16* __restrict__ Ct) {
    const int lane = threadIdx.x & 31;
    const int wid  = blockIdx.x * 8 + (threadIdx.x >> 5);  // 0..4095
    const int m0   = (wid >> 2) * 16;                      // M tile
    const int n0   = (wid & 3) * 64;                       // N block
    const int nn   = lane & 15, hi = lane >> 4;

    const __bf16* arow  = A  + (size_t)(m0 + nn) * HH;
    const __bf16* brow0 = Bt + (size_t)(n0 +  0 + nn) * HH;
    const __bf16* brow1 = Bt + (size_t)(n0 + 16 + nn) * HH;
    const __bf16* brow2 = Bt + (size_t)(n0 + 32 + nn) * HH;
    const __bf16* brow3 = Bt + (size_t)(n0 + 48 + nn) * HH;

    v8f acc0 = v8f_zero(), acc1 = v8f_zero(), acc2 = v8f_zero(), acc3 = v8f_zero();

    // software pipeline: fragments for step kb are loaded during step kb-32
    v16bf af  = load_a_frag(arow,  0, lane);
    v16bf bf0 = load_b_frag(brow0, 0, lane);
    v16bf bf1 = load_b_frag(brow1, 0, lane);
    v16bf bf2 = load_b_frag(brow2, 0, lane);
    v16bf bf3 = load_b_frag(brow3, 0, lane);

    for (int kb = 32; kb < HH; kb += 32) {
        v16bf afn = load_a_frag(arow,  kb, lane);
        v16bf b0n = load_b_frag(brow0, kb, lane);
        v16bf b1n = load_b_frag(brow1, kb, lane);
        v16bf b2n = load_b_frag(brow2, kb, lane);
        v16bf b3n = load_b_frag(brow3, kb, lane);
        acc0 = wmma_bf16(af, bf0, acc0);
        acc1 = wmma_bf16(af, bf1, acc1);
        acc2 = wmma_bf16(af, bf2, acc2);
        acc3 = wmma_bf16(af, bf3, acc3);
        af = afn; bf0 = b0n; bf1 = b1n; bf2 = b2n; bf3 = b3n;
    }
    acc0 = wmma_bf16(af, bf0, acc0);
    acc1 = wmma_bf16(af, bf1, acc1);
    acc2 = wmma_bf16(af, bf2, acc2);
    acc3 = wmma_bf16(af, bf3, acc3);

    v8f* accs[4] = {&acc0, &acc1, &acc2, &acc3};
    #pragma unroll
    for (int j = 0; j < 4; ++j) {
        #pragma unroll
        for (int r = 0; r < 8; ++r) {
            int m = m0 + r + 8 * hi;
            int n = n0 + 16 * j + nn;
            __bf16 val = f2bf((*accs[j])[r]);
            Cm[(size_t)m * DD + n] = val;
            if (Ct) {
                int b = m >> 10, s = m & (SS - 1);
                Ct[((size_t)b * DD + n) * SS + s] = val;
            }
        }
    }
}

// ---------------------------------------------------------------------------
// Output GEMM: Out[M=B*S, N=H] f32 = A[M, D] * Wo  (Wo given as Wo^T [H, D])
// ---------------------------------------------------------------------------
__global__ __launch_bounds__(256) void out_gemm(const __bf16* __restrict__ A,
                                                const __bf16* __restrict__ Bt,
                                                float* __restrict__ Out) {
    const int lane = threadIdx.x & 31;
    const int wid  = blockIdx.x * 8 + (threadIdx.x >> 5);  // 0..32767
    const int m0   = (wid >> 5) * 16;
    const int n0   = (wid & 31) * 64;
    const int nn   = lane & 15, hi = lane >> 4;

    const __bf16* arow  = A  + (size_t)(m0 + nn) * DD;
    const __bf16* brow0 = Bt + (size_t)(n0 +  0 + nn) * DD;
    const __bf16* brow1 = Bt + (size_t)(n0 + 16 + nn) * DD;
    const __bf16* brow2 = Bt + (size_t)(n0 + 32 + nn) * DD;
    const __bf16* brow3 = Bt + (size_t)(n0 + 48 + nn) * DD;

    v8f acc0 = v8f_zero(), acc1 = v8f_zero(), acc2 = v8f_zero(), acc3 = v8f_zero();

    v16bf af  = load_a_frag(arow,  0, lane);
    v16bf bf0 = load_b_frag(brow0, 0, lane);
    v16bf bf1 = load_b_frag(brow1, 0, lane);
    v16bf bf2 = load_b_frag(brow2, 0, lane);
    v16bf bf3 = load_b_frag(brow3, 0, lane);

    #pragma unroll
    for (int kb = 32; kb < DD; kb += 32) {
        v16bf afn = load_a_frag(arow,  kb, lane);
        v16bf b0n = load_b_frag(brow0, kb, lane);
        v16bf b1n = load_b_frag(brow1, kb, lane);
        v16bf b2n = load_b_frag(brow2, kb, lane);
        v16bf b3n = load_b_frag(brow3, kb, lane);
        acc0 = wmma_bf16(af, bf0, acc0);
        acc1 = wmma_bf16(af, bf1, acc1);
        acc2 = wmma_bf16(af, bf2, acc2);
        acc3 = wmma_bf16(af, bf3, acc3);
        af = afn; bf0 = b0n; bf1 = b1n; bf2 = b2n; bf3 = b3n;
    }
    acc0 = wmma_bf16(af, bf0, acc0);
    acc1 = wmma_bf16(af, bf1, acc1);
    acc2 = wmma_bf16(af, bf2, acc2);
    acc3 = wmma_bf16(af, bf3, acc3);

    v8f* accs[4] = {&acc0, &acc1, &acc2, &acc3};
    #pragma unroll
    for (int j = 0; j < 4; ++j)
        #pragma unroll
        for (int r = 0; r < 8; ++r)
            Out[(size_t)(m0 + r + 8 * hi) * HH + (n0 + 16 * j + nn)] = (*accs[j])[r];
}

// ---------------------------------------------------------------------------
// Flash-style attention:  out = softmax(Qm Km^T * scale [+ bias]) @ V
//   Qm  : [B, S, D] bf16 row-major  (queries, A-operand)
//   Km  : [B, S, D] bf16 row-major  (keys; rows feed B^T of score matmul)
//   VTm : [B, D, S] bf16            (values transposed; rows feed B of P@V)
//   bias: [S, S] f32 or nullptr     (async-DMA double-buffered into LDS)
// mode 0: outBf  = result                       (aggregation branches)
// mode 1: outAcc = w[branch]*result             (first triadic branch)
// mode 2: outAcc += w[branch]*result
// mode 3: outAcc += w[branch]*result ; outBf = bf16(outAcc)  (last branch)
// One wave handles 16 query rows.  grid = 128 blocks x 256 (8 waves/block).
// ---------------------------------------------------------------------------
__global__ __launch_bounds__(256) void attn_kernel(const __bf16* __restrict__ Qm,
                                                   const __bf16* __restrict__ Km,
                                                   const __bf16* __restrict__ VTm,
                                                   const float* __restrict__ bias,
                                                   float* __restrict__ outAcc,
                                                   __bf16* __restrict__ outBf,
                                                   const float* __restrict__ triadic,
                                                   int branch, int mode) {
    __shared__ __align__(16) __bf16 ldsP[8][16 * 32];
#if HAVE_ASYNC
    __shared__ __align__(16) float ldsBias[8][2][16 * 32];   // double-buffered
#endif

    const int lane = threadIdx.x & 31;
    const int wave = threadIdx.x >> 5;
    const int wid  = blockIdx.x * 8 + wave;   // 0..1023
    const int b    = wid >> 6;                // batch
    const int q0   = (wid & 63) * 16;         // query tile base
    const int nn   = lane & 15, hi = lane >> 4;
    const float scale = 0.0625f;              // D^-0.5
    const int NT = SS / 32;                   // 32 key tiles

    // Preload Q fragments for the whole D=256 contraction (8 k-steps)
    const __bf16* qrow = Qm + ((size_t)b * SS + q0 + nn) * DD;
    v16bf qf[8];
    #pragma unroll
    for (int j = 0; j < 8; ++j) qf[j] = load_a_frag(qrow, 32 * j, lane);

    v8f o[16];
    #pragma unroll
    for (int nt = 0; nt < 16; ++nt) o[nt] = v8f_zero();
    float mrow[8], lrow[8];
    #pragma unroll
    for (int r = 0; r < 8; ++r) { mrow[r] = -__builtin_inff(); lrow[r] = 0.0f; }

#if HAVE_ASYNC
    // Each tile: 4 wave-wide b128 async DMAs move the 16x32 f32 bias tile.
    const int br4  = lane >> 3;        // 0..3 (row within quad)
    const int bcol = (lane & 7) * 4;   // f32 column, 16B granules
    if (bias) {
        #pragma unroll
        for (int j = 0; j < 4; ++j) {
            int row = j * 4 + br4;
            async_g2l_b128(bias + (size_t)(q0 + row) * SS + 0 * 32 + bcol,
                           &ldsBias[wave][0][row * 32 + bcol]);
        }
    }
#endif

    for (int it = 0; it < NT; ++it) {
        const int kb = it * 32;
        const __bf16* krow0 = Km + ((size_t)b * SS + kb + nn) * DD;
        const __bf16* krow1 = Km + ((size_t)b * SS + kb + 16 + nn) * DD;
        if (kb + 32 < SS)   // hint next key tile into cache (global_prefetch_b8)
            __builtin_prefetch(Km + ((size_t)b * SS + kb + 32 + nn) * DD, 0, 1);

        // --- scores: two 16x16 C tiles covering 32 keys (pipelined loads) ---
        v8f c0 = v8f_zero(), c1 = v8f_zero();
        {
            v16bf kf0 = load_b_frag(krow0, 0, lane);
            v16bf kf1 = load_b_frag(krow1, 0, lane);
            #pragma unroll
            for (int j = 0; j < 8; ++j) {
                v16bf n0f = kf0, n1f = kf1;
                if (j < 7) {
                    n0f = load_b_frag(krow0, 32 * (j + 1), lane);
                    n1f = load_b_frag(krow1, 32 * (j + 1), lane);
                }
                c0 = wmma_bf16(qf[j], kf0, c0);
                c1 = wmma_bf16(qf[j], kf1, c1);
                kf0 = n0f; kf1 = n1f;
            }
        }

        // --- bias tile: async double buffer (or direct global fallback) ---
        float s0[8], s1[8];
        if (bias) {
#if HAVE_ASYNC
            if (it + 1 < NT) {
                #pragma unroll
                for (int j = 0; j < 4; ++j) {
                    int row = j * 4 + br4;
                    async_g2l_b128(bias + (size_t)(q0 + row) * SS + (kb + 32) + bcol,
                                   &ldsBias[wave][(it + 1) & 1][row * 32 + bcol]);
                }
                asm volatile("s_wait_asynccnt 0x4" ::: "memory");
            } else {
                asm volatile("s_wait_asynccnt 0x0" ::: "memory");
            }
            const float* bt = ldsBias[wave][it & 1];
            #pragma unroll
            for (int r = 0; r < 8; ++r) {
                int m = r + 8 * hi;
                s0[r] = c0[r] * scale + bt[m * 32 + nn];
                s1[r] = c1[r] * scale + bt[m * 32 + 16 + nn];
            }
#else
            #pragma unroll
            for (int r = 0; r < 8; ++r) {
                int m = q0 + r + 8 * hi;
                s0[r] = c0[r] * scale + bias[(size_t)m * SS + kb + nn];
                s1[r] = c1[r] * scale + bias[(size_t)m * SS + kb + 16 + nn];
            }
#endif
        } else {
            #pragma unroll
            for (int r = 0; r < 8; ++r) {
                s0[r] = c0[r] * scale;
                s1[r] = c1[r] * scale;
            }
        }

        // --- online softmax (rows live on 16 lanes; xor-reduce within half) ---
        float mnew[8], alpha[8], p0[8], p1[8];
        #pragma unroll
        for (int r = 0; r < 8; ++r) {
            float mx = fmaxf(s0[r], s1[r]);
            #pragma unroll
            for (int off = 1; off < 16; off <<= 1)
                mx = fmaxf(mx, __shfl_xor(mx, off, 32));
            mnew[r]  = fmaxf(mrow[r], mx);
            alpha[r] = __expf(mrow[r] - mnew[r]);
            mrow[r]  = mnew[r];
        }
        #pragma unroll
        for (int r = 0; r < 8; ++r) {
            p0[r] = __expf(s0[r] - mnew[r]);
            p1[r] = __expf(s1[r] - mnew[r]);
            float rs = p0[r] + p1[r];
            #pragma unroll
            for (int off = 1; off < 16; off <<= 1)
                rs += __shfl_xor(rs, off, 32);
            lrow[r] = lrow[r] * alpha[r] + rs;
        }
        #pragma unroll
        for (int nt = 0; nt < 16; ++nt)
            #pragma unroll
            for (int r = 0; r < 8; ++r) o[nt][r] *= alpha[r];

        // --- C-layout P tile -> LDS -> A-layout fragment ---
        __bf16* pl = ldsP[wave];
        #pragma unroll
        for (int r = 0; r < 8; ++r) {
            int m = r + 8 * hi;
            pl[m * 32 + nn]      = f2bf(p0[r]);
            pl[m * 32 + 16 + nn] = f2bf(p1[r]);
        }
        asm volatile("s_wait_dscnt 0" ::: "memory");
        v16bf pf = load_a_frag(pl + nn * 32, 0, lane);

        // --- O += P @ V over all 16 d-tiles (pipelined V fragment loads) ---
        const __bf16* vbase = VTm + (size_t)b * DD * SS;
        v16bf vf = load_b_frag(vbase + (size_t)(0 + nn) * SS, kb, lane);
        #pragma unroll
        for (int nt = 0; nt < 16; ++nt) {
            v16bf nvf = vf;
            if (nt < 15)
                nvf = load_b_frag(vbase + (size_t)((nt + 1) * 16 + nn) * SS, kb, lane);
            o[nt] = wmma_bf16(pf, vf, o[nt]);
            vf = nvf;
        }
    }

    // --- epilogue: normalize, triadic weight, write ---
    float wscale = 1.0f;
    if (triadic) {
        float t0 = triadic[0], t1 = triadic[1], t2 = triadic[2];
        float mx = fmaxf(t0, fmaxf(t1, t2));
        float e0 = __expf(t0 - mx), e1 = __expf(t1 - mx), e2 = __expf(t2 - mx);
        float sel = (branch == 0) ? e0 : (branch == 1) ? e1 : e2;
        wscale = sel / (e0 + e1 + e2);
    }
    float rl[8];
    #pragma unroll
    for (int r = 0; r < 8; ++r) rl[r] = 1.0f / lrow[r];

    #pragma unroll
    for (int nt = 0; nt < 16; ++nt) {
        #pragma unroll
        for (int r = 0; r < 8; ++r) {
            int m = q0 + r + 8 * hi;
            int d = nt * 16 + nn;
            size_t idx = ((size_t)b * SS + m) * DD + d;
            float val = o[nt][r] * rl[r];
            if (mode == 0) {
                outBf[idx] = f2bf(val);
            } else if (mode == 1) {
                outAcc[idx] = wscale * val;
            } else if (mode == 2) {
                outAcc[idx] += wscale * val;
            } else {
                float s = outAcc[idx] + wscale * val;
                outAcc[idx] = s;
                outBf[idx] = f2bf(s);
            }
        }
    }
}

// ---------------------------------------------------------------------------
// Host launcher
// ---------------------------------------------------------------------------
extern "C" void kernel_launch(void* const* d_in, const int* in_sizes, int n_in,
                              void* d_out, int out_size, void* d_ws, size_t ws_size,
                              hipStream_t stream) {
    const float* hs      = (const float*)d_in[0];  // [B,S,H]
    const float* Wq      = (const float*)d_in[1];  // [H,D]
    const float* Wk      = (const float*)d_in[2];
    const float* Wk2     = (const float*)d_in[3];
    const float* Wv      = (const float*)d_in[4];
    const float* Wo      = (const float*)d_in[5];  // [D,H]
    const float* triadic = (const float*)d_in[6];  // [3]
    const float* pbias   = (const float*)d_in[7];  // [S,S]
    float* out = (float*)d_out;                    // [B,S,H]

    char* ws = (char*)d_ws;
    size_t off = 0;
    auto carve = [&](size_t bytes) -> char* {
        char* p = ws + off;
        off += (bytes + 255) & ~(size_t)255;
        return p;
    };
    const size_t BSH = (size_t)BB * SS * HH;   // 33.5M
    const size_t BSD = (size_t)BB * SS * DD;   // 4.19M
    const size_t HD  = (size_t)HH * DD;

    __bf16* hsb   = (__bf16*)carve(BSH * 2);
    __bf16* wqT   = (__bf16*)carve(HD * 2);
    __bf16* wkT   = (__bf16*)carve(HD * 2);
    __bf16* wk2T  = (__bf16*)carve(HD * 2);
    __bf16* wvT   = (__bf16*)carve(HD * 2);
    __bf16* woT   = (__bf16*)carve(HD * 2);
    __bf16* Qb    = (__bf16*)carve(BSD * 2);
    __bf16* Kb    = (__bf16*)carve(BSD * 2);
    __bf16* K2b   = (__bf16*)carve(BSD * 2);
    __bf16* Vb    = (__bf16*)carve(BSD * 2);
    __bf16* KT    = (__bf16*)carve(BSD * 2);
    __bf16* K2T   = (__bf16*)carve(BSD * 2);
    __bf16* VT    = (__bf16*)carve(BSD * 2);
    __bf16* Kagg  = (__bf16*)carve(BSD * 2);
    __bf16* K2agg = (__bf16*)carve(BSD * 2);
    float*  combF = (float*)carve(BSD * 4);
    __bf16* combB = (__bf16*)carve(BSD * 2);

    // 1) casts / weight transposes
    cast_f32_bf16<<<(int)((BSH + 255) / 256), 256, 0, stream>>>(hs, hsb, (int)BSH);
    transpose_cast<<<(int)((HD + 255) / 256), 256, 0, stream>>>(Wq,  wqT,  HH, DD);
    transpose_cast<<<(int)((HD + 255) / 256), 256, 0, stream>>>(Wk,  wkT,  HH, DD);
    transpose_cast<<<(int)((HD + 255) / 256), 256, 0, stream>>>(Wk2, wk2T, HH, DD);
    transpose_cast<<<(int)((HD + 255) / 256), 256, 0, stream>>>(Wv,  wvT,  HH, DD);
    transpose_cast<<<(int)((HD + 255) / 256), 256, 0, stream>>>(Wo,  woT,  DD, HH);

    // 2) projections (Q plain; K/K2/V also transposed per batch)
    proj_gemm<<<512, 256, 0, stream>>>(hsb, wqT,  Qb,  (__bf16*)nullptr);
    proj_gemm<<<512, 256, 0, stream>>>(hsb, wkT,  Kb,  KT);
    proj_gemm<<<512, 256, 0, stream>>>(hsb, wk2T, K2b, K2T);
    proj_gemm<<<512, 256, 0, stream>>>(hsb, wvT,  Vb,  VT);

    // 3) jump aggregations: K_agg = softmax(K K2^T s) K2 ; K2_agg = softmax(K2 K^T s) K
    attn_kernel<<<128, 256, 0, stream>>>(Kb,  K2b, K2T, (const float*)nullptr,
                                         (float*)nullptr, Kagg, (const float*)nullptr, 0, 0);
    attn_kernel<<<128, 256, 0, stream>>>(K2b, Kb,  KT,  (const float*)nullptr,
                                         (float*)nullptr, K2agg, (const float*)nullptr, 0, 0);

    // 4) three biased attention branches fused into weighted accumulator
    attn_kernel<<<128, 256, 0, stream>>>(Qb, Kb,    VT, pbias, combF, (__bf16*)nullptr, triadic, 0, 1);
    attn_kernel<<<128, 256, 0, stream>>>(Qb, Kagg,  VT, pbias, combF, (__bf16*)nullptr, triadic, 1, 2);
    attn_kernel<<<128, 256, 0, stream>>>(Qb, K2agg, VT, pbias, combF, combB,            triadic, 2, 3);

    // 5) output projection -> d_out f32 [B,S,H]
    out_gemm<<<4096, 256, 0, stream>>>(combB, woT, out);
}
